// Attention_64063732187236
// MI455X (gfx1250) — matrix operand order repaired and checked by reading, not compiled, not run on hardware
//
#include <hip/hip_runtime.h>

typedef __attribute__((ext_vector_type(16))) _Float16 v16h;
typedef __attribute__((ext_vector_type(8)))  _Float16 v8h;
typedef __attribute__((ext_vector_type(4)))  _Float16 v4h;
typedef __attribute__((ext_vector_type(8)))  float    v8f;

#define NTHREADS 128

constexpr int B_ = 4, N_ = 2048, DIM_ = 512, HEADS_ = 8, DHEAD_ = 64, INNER_ = 512;
constexpr float QSCALE = 0.35355339059327373f;   // HEADS^-0.5 (faithful source quirk)
constexpr float NEGBIG = -1.0e12f;

// ---------------------------------------------------------------------------
// CDNA5 async global->LDS copy (GLOBAL_LOAD_ASYNC_TO_LDS_B128, ASYNCcnt)
// LDS dest address = low 32 bits of the flat pointer (ISA 10.2 aperture rule).
// ---------------------------------------------------------------------------
__device__ inline void wait_async0() {
#if __has_builtin(__builtin_amdgcn_s_wait_asynccnt)
  __builtin_amdgcn_s_wait_asynccnt(0);
#else
  asm volatile("s_wait_asynccnt 0x0" ::: "memory");
#endif
}

template <int ROWS, int COLS_BYTES, int LDD_BYTES>
__device__ inline void stage_async(const void* __restrict__ src, int ld_src_bytes,
                                   void* dst, int tid) {
  constexpr int ITERS = ROWS * COLS_BYTES / 16 / NTHREADS;
#pragma unroll
  for (int it = 0; it < ITERS; ++it) {
    const int i = tid + it * NTHREADS;
    const int r = (i * 16) / COLS_BYTES;
    const int c = (i * 16) % COLS_BYTES;
    const unsigned lds = (unsigned)(uintptr_t)((char*)dst + r * LDD_BYTES + c);
    const unsigned long long g =
        (unsigned long long)(uintptr_t)((const char*)src + (size_t)r * ld_src_bytes + c);
    asm volatile("global_load_async_to_lds_b128 %0, %1, off"
                 :: "v"(lds), "v"(g) : "memory");
  }
}

// ---------------------------------------------------------------------------
// WMMA fragment loaders — per-lane reads are 2 x 16B contiguous (ds_load_b128)
// ---------------------------------------------------------------------------
__device__ inline v16h ld_frag_a(const _Float16* lds, int ld) {
  const int lane = threadIdx.x & 31;
  const _Float16* p = lds + (lane & 15) * ld + ((lane >> 4) << 3);
  const v8h lo = *(const v8h*)__builtin_assume_aligned(p, 16);
  const v8h hi = *(const v8h*)__builtin_assume_aligned(p + 16, 16);
  return __builtin_shufflevector(lo, hi, 0, 1, 2, 3, 4, 5, 6, 7,
                                 8, 9, 10, 11, 12, 13, 14, 15);
}

// B-matrix 32x16 (KxN), column-major storage: element (k,c) at lds[c*ld+k].
__device__ inline v16h ld_frag_b_cm(const _Float16* lds, int ld) {
  const int lane = threadIdx.x & 31;
  const _Float16* p = lds + (lane & 15) * ld + ((lane >> 4) << 3);
  const v8h lo = *(const v8h*)__builtin_assume_aligned(p, 16);
  const v8h hi = *(const v8h*)__builtin_assume_aligned(p + 16, 16);
  return __builtin_shufflevector(lo, hi, 0, 1, 2, 3, 4, 5, 6, 7,
                                 8, 9, 10, 11, 12, 13, 14, 15);
}

__device__ inline v8f wmma_f16(v16h a, v16h b, v8f c) {
  return __builtin_amdgcn_wmma_f32_16x16x32_f16(false, a, false, b, (short)0, c,
                                                false, false);
}

// ---------------------------------------------------------------------------
// Prologue kernels: fp32 -> f16 copies so the GEMM hot loops are pure
// async byte copies.  Weights stored pre-transposed [n][k].
// ---------------------------------------------------------------------------
__global__ __launch_bounds__(256)
void cvt_f16_kernel(const float* __restrict__ src, _Float16* __restrict__ dst) {
  const size_t i = ((size_t)blockIdx.x * 256 + threadIdx.x) * 4;
  const float4 v = *(const float4*)(src + i);
  v4h q = {(_Float16)v.x, (_Float16)v.y, (_Float16)v.z, (_Float16)v.w};
  *(v4h*)(dst + i) = q;
}

template <int K, int NCOL>   // src [K][NCOL] f32 -> dst [NCOL][K] f16
__global__ __launch_bounds__(256)
void transpose_f16_kernel(const float* __restrict__ src, _Float16* __restrict__ dst) {
  const size_t idx = (size_t)blockIdx.x * 256 + threadIdx.x;
  const int k = (int)(idx / NCOL);
  const int n = (int)(idx % NCOL);
  dst[(size_t)n * K + k] = (_Float16)src[idx];
}

// ---------------------------------------------------------------------------
// Shared GEMM core: C[64 x 128] += A[64 x 512] x B[512 x 128], all tiles
// async-staged f16, double-buffered.  B source is pre-transposed [n][k].
// Per k-step: all 8 B fragments issued before the WMMA chain.
// ---------------------------------------------------------------------------
constexpr int LDA = 40, LDB = 40;

template <int KTOT>
__device__ inline void gemm_core(const _Float16* __restrict__ Arow,
                                 const _Float16* __restrict__ Brow,
                                 _Float16 (&As)[2][64 * LDA],
                                 _Float16 (&BsT)[2][128 * LDB],
                                 v8f (&acc)[8], int tid, int w) {
  stage_async<64, 64, LDA * 2>(Arow, KTOT * 2, As[0], tid);
  stage_async<128, 64, LDB * 2>(Brow, KTOT * 2, BsT[0], tid);
  wait_async0();
  __syncthreads();

  constexpr int NSTEP = KTOT / 32;
  for (int k = 0; k < NSTEP; ++k) {
    const int cur = k & 1;
    if (k + 1 < NSTEP) {        // async-stage next chunk; overlaps compute
      const int kc = (k + 1) * 32;
      stage_async<64, 64, LDA * 2>(Arow + kc, KTOT * 2, As[cur ^ 1], tid);
      stage_async<128, 64, LDB * 2>(Brow + kc, KTOT * 2, BsT[cur ^ 1], tid);
    }
    v16h bf[8];
#pragma unroll
    for (int j = 0; j < 8; ++j)
      bf[j] = ld_frag_b_cm(BsT[cur] + (j * 16) * LDB, LDB);
    const v16h a = ld_frag_a(As[cur] + (w * 16) * LDA, LDA);
#pragma unroll
    for (int j = 0; j < 8; ++j) acc[j] = wmma_f16(a, bf[j], acc[j]);
    wait_async0();
    __syncthreads();
  }
}

// ---------------------------------------------------------------------------
// Kernel 1: QKV projection  Xh[8192x512]f16 x WqkvT[1536x512]f16
// -> Q,K f16 [b][h][n][d] (Q pre-scaled), V transposed [b][h][d][n].
// ---------------------------------------------------------------------------
__global__ __launch_bounds__(NTHREADS)
void qkv_proj_kernel(const _Float16* __restrict__ Xh, const _Float16* __restrict__ WqkvT,
                     _Float16* __restrict__ Qb, _Float16* __restrict__ Kb,
                     _Float16* __restrict__ Vtb) {
  __shared__ alignas(16) _Float16 As[2][64 * LDA];    // [m][k]
  __shared__ alignas(16) _Float16 BsT[2][128 * LDB];  // [n][k]

  const int n0   = blockIdx.x * 128;
  const int m0   = blockIdx.y * 64;
  const int tid  = threadIdx.x;
  const int w    = tid >> 5;
  const int lane = tid & 31;

  v8f acc[8] = {};
  gemm_core<DIM_>(Xh + (size_t)m0 * DIM_, WqkvT + (size_t)n0 * DIM_, As, BsT,
                  acc, tid, w);

  const int col16 = lane & 15;
  const int rb    = (lane >> 4) * 8;
  const int gm0   = m0 + w * 16 + rb;
  const int b     = gm0 >> 11;
  const int nr0   = gm0 & 2047;
#pragma unroll
  for (int j = 0; j < 8; ++j) {
    const int gn      = n0 + j * 16 + col16;
    const int section = gn >> 9;        // 0=Q 1=K 2=V (uniform per fragment)
    const int cc      = gn & 511;
    const int h       = cc >> 6;
    const int d       = cc & 63;
    if (section == 2) {
      v8h pk;
#pragma unroll
      for (int i = 0; i < 8; ++i) pk[i] = (_Float16)acc[j][i];
      *(v8h*)(Vtb + (((size_t)b * HEADS_ + h) * DHEAD_ + d) * N_ + nr0) = pk;
    } else {
      _Float16* dst  = (section == 0) ? Qb : Kb;
      const float sc = (section == 0) ? QSCALE : 1.0f;
#pragma unroll
      for (int i = 0; i < 8; ++i)
        dst[(((size_t)b * HEADS_ + h) * N_ + nr0 + i) * DHEAD_ + d] =
            (_Float16)(acc[j][i] * sc);
    }
  }
}

// ---------------------------------------------------------------------------
// Kernel 2: fused flash attention per (b,h) with DropKey mask.
// grid.x = query strip (N/64), grid.y = b*8+h.  4 waves, 16 query rows/wave.
// K/V/mask tiles async-staged (ping-pong) for tile kt+1 during compute of kt.
// ---------------------------------------------------------------------------
__global__ __launch_bounds__(NTHREADS)
void flash_attn_kernel(const _Float16* __restrict__ Qb, const _Float16* __restrict__ Kb,
                       const _Float16* __restrict__ Vtb,
                       const unsigned char* __restrict__ Mask,
                       _Float16* __restrict__ Ab) {
  constexpr int LD  = 72;
  constexpr int LDBY = LD * (int)sizeof(_Float16);   // 144 bytes, 16B multiple
  __shared__ alignas(16) _Float16 Qs[64 * LD];        // [q][d]
  __shared__ alignas(16) _Float16 Ks[2][64 * LD];     // [key][d]
  __shared__ alignas(16) _Float16 Vs[2][64 * LD];     // [d][key]
  __shared__ alignas(16) _Float16 Ps[64 * LD];        // per-wave [q][key]
  __shared__ alignas(16) unsigned char Ms[2][64 * 64];

  const int tid  = threadIdx.x;
  const int w    = tid >> 5;
  const int lane = tid & 31;
  const int bh   = blockIdx.y;
  const int q0   = blockIdx.x * 64;
  const size_t head_off = (size_t)bh * N_ * DHEAD_;
  const unsigned char* mrow = Mask + ((size_t)bh * N_ + q0) * N_;

  stage_async<64, 128, LDBY>(Qb + head_off + (size_t)q0 * DHEAD_, DHEAD_ * 2, Qs, tid);
  stage_async<64, 128, LDBY>(Kb + head_off, DHEAD_ * 2, Ks[0], tid);
  stage_async<64, 128, LDBY>(Vtb + head_off, N_ * 2, Vs[0], tid);
  stage_async<64, 64, 64>(mrow, N_, Ms[0], tid);
  wait_async0();
  __syncthreads();

  const v16h qa0 = ld_frag_a(Qs + (w * 16) * LD + 0, LD);
  const v16h qa1 = ld_frag_a(Qs + (w * 16) * LD + 32, LD);

  v8f o[4] = {};
  float m8[8], l8[8];
#pragma unroll
  for (int i = 0; i < 8; ++i) { m8[i] = -3.0e38f; l8[i] = 0.0f; }

  const int col16 = lane & 15;
  const int rb    = (lane >> 4) * 8;

  for (int kt = 0; kt < N_ / 64; ++kt) {
    const int cur = kt & 1;
    if (kt + 1 < N_ / 64) {   // async-stage next tile; overlaps compute below
      const int kn = (kt + 1) * 64;
      stage_async<64, 128, LDBY>(Kb + head_off + (size_t)kn * DHEAD_, DHEAD_ * 2,
                                 Ks[cur ^ 1], tid);
      stage_async<64, 128, LDBY>(Vtb + head_off + kn, N_ * 2, Vs[cur ^ 1], tid);
      stage_async<64, 64, 64>(mrow + kn, N_, Ms[cur ^ 1], tid);
    }

    // S = Q x K^T : all 8 K fragments issued before the WMMA chain
    v16h kb[8];
#pragma unroll
    for (int j = 0; j < 4; ++j) {
      kb[2 * j]     = ld_frag_b_cm(Ks[cur] + (j * 16) * LD + 0, LD);
      kb[2 * j + 1] = ld_frag_b_cm(Ks[cur] + (j * 16) * LD + 32, LD);
    }
    v8f s[4];
#pragma unroll
    for (int j = 0; j < 4; ++j) {
      v8f c = {};
      c = wmma_f16(qa0, kb[2 * j], c);
      c = wmma_f16(qa1, kb[2 * j + 1], c);
      s[j] = c;
    }

    // DropKey: atten += mask * NEG_BIG
#pragma unroll
    for (int j = 0; j < 4; ++j)
#pragma unroll
      for (int i = 0; i < 8; ++i) {
        const int r = w * 16 + rb + i;
        s[j][i] = fmaf((float)Ms[cur][r * 64 + j * 16 + col16], NEGBIG, s[j][i]);
      }

    // online softmax (row stats via xor-shuffles within 16-lane groups)
    float mnew[8];
#pragma unroll
    for (int i = 0; i < 8; ++i) {
      float v = fmaxf(fmaxf(s[0][i], s[1][i]), fmaxf(s[2][i], s[3][i]));
      v = fmaxf(v, __shfl_xor(v, 1));
      v = fmaxf(v, __shfl_xor(v, 2));
      v = fmaxf(v, __shfl_xor(v, 4));
      v = fmaxf(v, __shfl_xor(v, 8));
      mnew[i] = fmaxf(m8[i], v);
    }
#pragma unroll
    for (int j = 0; j < 4; ++j)
#pragma unroll
      for (int i = 0; i < 8; ++i) s[j][i] = __expf(s[j][i] - mnew[i]);
#pragma unroll
    for (int i = 0; i < 8; ++i) {
      float v = (s[0][i] + s[1][i]) + (s[2][i] + s[3][i]);
      v += __shfl_xor(v, 1);
      v += __shfl_xor(v, 2);
      v += __shfl_xor(v, 4);
      v += __shfl_xor(v, 8);
      const float alpha = __expf(m8[i] - mnew[i]);
      l8[i] = l8[i] * alpha + v;
      m8[i] = mnew[i];
#pragma unroll
      for (int j = 0; j < 4; ++j) o[j][i] *= alpha;
    }

    // V fragments first (independent), then P LDS round-trip (in-order DS
    // guarantees the P loads observe the stores), then the PV WMMA chain.
    v16h vb[8];
#pragma unroll
    for (int dj = 0; dj < 4; ++dj) {
      vb[2 * dj]     = ld_frag_b_cm(Vs[cur] + (dj * 16) * LD + 0, LD);
      vb[2 * dj + 1] = ld_frag_b_cm(Vs[cur] + (dj * 16) * LD + 32, LD);
    }
#pragma unroll
    for (int j = 0; j < 4; ++j)
#pragma unroll
      for (int i = 0; i < 8; ++i)
        Ps[(w * 16 + rb + i) * LD + j * 16 + col16] = (_Float16)s[j][i];

    const v16h pa0 = ld_frag_a(Ps + (w * 16) * LD + 0, LD);
    const v16h pa1 = ld_frag_a(Ps + (w * 16) * LD + 32, LD);
#pragma unroll
    for (int dj = 0; dj < 4; ++dj) {
      o[dj] = wmma_f16(pa0, vb[2 * dj], o[dj]);
      o[dj] = wmma_f16(pa1, vb[2 * dj + 1], o[dj]);
    }

    wait_async0();     // next tile resident before anyone proceeds
    __syncthreads();
  }

  // normalize, write attn output re-assembled as [B][N][H*D] f16
  const int b = bh >> 3, h = bh & 7;
#pragma unroll
  for (int i = 0; i < 8; ++i) {
    const float inv = 1.0f / l8[i];
    const int qrow  = q0 + w * 16 + rb + i;
    _Float16* dst = Ab + ((size_t)b * N_ + qrow) * INNER_ + h * DHEAD_;
#pragma unroll
    for (int dj = 0; dj < 4; ++dj)
      dst[dj * 16 + col16] = (_Float16)(o[dj][i] * inv);
  }
}

// ---------------------------------------------------------------------------
// Kernel 3: output projection  Ab[8192x512]f16 x WoutT[512x512]f16 -> f32
// ---------------------------------------------------------------------------
__global__ __launch_bounds__(NTHREADS)
void out_proj_kernel(const _Float16* __restrict__ A, const _Float16* __restrict__ WoutT,
                     float* __restrict__ Out) {
  __shared__ alignas(16) _Float16 As[2][64 * LDA];
  __shared__ alignas(16) _Float16 BsT[2][128 * LDB];

  const int n0   = blockIdx.x * 128;
  const int m0   = blockIdx.y * 64;
  const int tid  = threadIdx.x;
  const int w    = tid >> 5;
  const int lane = tid & 31;

  v8f acc[8] = {};
  gemm_core<INNER_>(A + (size_t)m0 * INNER_, WoutT + (size_t)n0 * INNER_, As, BsT,
                    acc, tid, w);

  const int col16 = lane & 15;
  const int rb    = (lane >> 4) * 8;
#pragma unroll
  for (int j = 0; j < 8; ++j)
#pragma unroll
    for (int i = 0; i < 8; ++i)
      Out[(size_t)(m0 + w * 16 + rb + i) * DIM_ + n0 + j * 16 + col16] = acc[j][i];
}

// ---------------------------------------------------------------------------
extern "C" void kernel_launch(void* const* d_in, const int* in_sizes, int n_in,
                              void* d_out, int out_size, void* d_ws, size_t ws_size,
                              hipStream_t stream) {
  const float*         x     = (const float*)d_in[0];
  const unsigned char* mask  = (const unsigned char*)d_in[1];  // jnp bool = 1 byte
  const float*         w_qkv = (const float*)d_in[2];
  const float*         w_out = (const float*)d_in[3];
  float*               out   = (float*)d_out;

  // workspace (f16): Q,K [b][h][n][d], Vt [b][h][d][n], attn-out, Xh, WqkvT, WoutT
  const size_t headElems = (size_t)B_ * HEADS_ * N_ * DHEAD_;  // 4,194,304
  _Float16* Qb    = (_Float16*)d_ws;
  _Float16* Kb    = Qb + headElems;
  _Float16* Vtb   = Kb + headElems;
  _Float16* Ab    = Vtb + headElems;
  _Float16* Xh    = Ab + headElems;                  // 8192*512
  _Float16* WqkvT = Xh + (size_t)B_ * N_ * DIM_;     // 1536*512
  _Float16* WoutT = WqkvT + (size_t)(3 * INNER_) * DIM_;  // 512*512  (total 42 MB)

  dim3 blk(NTHREADS);
  cvt_f16_kernel<<<(B_ * N_ * DIM_) / (256 * 4), 256, 0, stream>>>(x, Xh);
  transpose_f16_kernel<DIM_, 3 * INNER_>
      <<<(DIM_ * 3 * INNER_) / 256, 256, 0, stream>>>(w_qkv, WqkvT);
  transpose_f16_kernel<INNER_, DIM_>
      <<<(INNER_ * DIM_) / 256, 256, 0, stream>>>(w_out, WoutT);

  qkv_proj_kernel<<<dim3(3 * INNER_ / 128, (B_ * N_) / 64), blk, 0, stream>>>(
      Xh, WqkvT, Qb, Kb, Vtb);
  flash_attn_kernel<<<dim3(N_ / 64, B_ * HEADS_), blk, 0, stream>>>(
      Qb, Kb, Vtb, mask, Ab);
  out_proj_kernel<<<dim3(DIM_ / 128, (B_ * N_) / 64), blk, 0, stream>>>(
      Ab, WoutT, out);
}
